// my_MaskedMHA_36524401885768
// MI455X (gfx1250) — matrix operand order repaired and checked by reading, not compiled, not run on hardware
//
#include <hip/hip_runtime.h>
#include <hip/hip_bf16.h>
#include <math.h>
#include <stdint.h>

// ---------------------------------------------------------------------------
// Problem constants (match reference): B=2, S=2048, D=1024, H=16, DH=64
// ---------------------------------------------------------------------------
#define B_   2
#define S_   2048
#define D_   1024
#define H_   16
#define DH_  64
#define BH_  (B_ * H_)

typedef __attribute__((ext_vector_type(16))) __bf16 v16bf;
typedef __attribute__((ext_vector_type(8)))  float  v8f;

// ---------------------------------------------------------------------------
// CDNA5 async copy: global -> LDS without touching VGPR data path.
// the LDS address is taken from the VDST VGPR; flat shared pointers carry the group-segment
// byte offset in their low 32 bits (aperture tag lives in [63:32]).
// Tracked by ASYNCcnt -> s_wait_asynccnt.
// ---------------------------------------------------------------------------
__device__ __forceinline__ void async_b128(__bf16* lds_dst, const __bf16* gsrc) {
#if defined(__gfx1250__)
    const unsigned lds_off         = (unsigned)(uintptr_t)lds_dst;
    const unsigned long long gaddr = (unsigned long long)(uintptr_t)gsrc;
    asm volatile("global_load_async_to_lds_b128 %0, %1, off"
                 :: "v"(lds_off), "v"(gaddr) : "memory");
#else
    for (int i = 0; i < 8; ++i) lds_dst[i] = gsrc[i];
#endif
}

__device__ __forceinline__ void wait_async0() {
#if defined(__gfx1250__)
    asm volatile("s_wait_asynccnt 0" ::: "memory");
#endif
}

// ---------------------------------------------------------------------------
// WMMA fragment loaders (wave32, v_wmma_f32_16x16x32_bf16; this code's contiguous k = 16*(L>>4)+s, same for A and B).
// A 16x32 (MxK): lane L: m=L&15, kb=(L>>4)*16; VGPR0..7 = K kb..kb+15
//   -> one contiguous 32B run per lane.
// B 32x16 (KxN) stored N-major in LDS: lane L: n=L&15, kb=(L>>4)*16;
//   VGPR0..7 = K kb..kb+15     -> one contiguous 32B run per lane.
// STRIDE = LDS row pitch in elements.
// ---------------------------------------------------------------------------
template <int STRIDE>
__device__ __forceinline__ v16bf load_a_frag(const __bf16* lds, int mBase, int lane) {
    const int m  = mBase + (lane & 15);
    const int kb = (lane >> 4) * 16;
    const __bf16* p0 = lds + m * STRIDE + kb;
    const __bf16* p1 = p0 + 8;
    v16bf r;
#pragma unroll
    for (int i = 0; i < 8; ++i) { r[i] = p0[i]; r[8 + i] = p1[i]; }
    return r;
}

template <int STRIDE>
__device__ __forceinline__ v16bf load_b_frag(const __bf16* lds, int nBase, int lane) {
    const int n  = nBase + (lane & 15);
    const int kb = (lane >> 4) * 16;
    const __bf16* p = lds + n * STRIDE + kb;
    v16bf r;
#pragma unroll
    for (int i = 0; i < 16; ++i) r[i] = p[i];
    return r;
}

#define WMMA_BF16(a, b, c) \
    __builtin_amdgcn_wmma_f32_16x16x32_bf16(false, (a), false, (b), (short)0, (c), false, false)

// ---------------------------------------------------------------------------
// Kernel 1/6: C[M,N] = A[M,K] @ W[N,K]^T + bias  (f32 io, bf16 WMMA, f32 acc)
// 256 thr / 8 waves; tile 128x128, K-slab 32, LDS double-buffered so the
// f32 loads + cvt_pk_bf16 of slab i+1 overlap the WMMAs of slab i.
// Wave grid 4x2: each wave owns 32(M) x 64(N) = 2x4 wmma tiles.
// ---------------------------------------------------------------------------
__global__ __launch_bounds__(256) void gemm_xw_bias(
    const float* __restrict__ A, const float* __restrict__ W,
    const float* __restrict__ bias, float* __restrict__ C,
    int M, int N, int K)
{
    __shared__ __bf16 lA[2][128 * 32];
    __shared__ __bf16 lB[2][128 * 32];

    const int tid  = threadIdx.x;
    const int lane = tid & 31;
    const int w    = tid >> 5;
    const int wr   = w >> 1;      // 0..3
    const int wc   = w & 1;       // 0..1
    const int mTile = blockIdx.y * 128;
    const int nTile = blockIdx.x * 128;

    v8f acc[2][4];
#pragma unroll
    for (int i = 0; i < 2; ++i)
#pragma unroll
        for (int j = 0; j < 4; ++j)
#pragma unroll
            for (int e = 0; e < 8; ++e) acc[i][j][e] = 0.f;

    const int row  = tid >> 1;          // 0..127
    const int colb = (tid & 1) * 16;    // 0 or 16
    const float* srcA0 = A + (size_t)(mTile + row) * K + colb;
    const float* srcW0 = W + (size_t)(nTile + row) * K + colb;

    // prologue: stage slab 0 into buffer 0
    {
        __bf16* dA = &lA[0][row * 32 + colb];
        __bf16* dB = &lB[0][row * 32 + colb];
#pragma unroll
        for (int i = 0; i < 16; ++i) { dA[i] = (__bf16)srcA0[i]; dB[i] = (__bf16)srcW0[i]; }
    }
    __syncthreads();

    int buf = 0;
    for (int k0 = 0; k0 < K; k0 += 32, buf ^= 1) {
        // stage slab k0+32 into the other buffer (overlaps WMMAs below)
        if (k0 + 32 < K) {
            const float* sA = srcA0 + k0 + 32;
            const float* sW = srcW0 + k0 + 32;
            __bf16* dA = &lA[buf ^ 1][row * 32 + colb];
            __bf16* dB = &lB[buf ^ 1][row * 32 + colb];
#pragma unroll
            for (int i = 0; i < 16; ++i) { dA[i] = (__bf16)sA[i]; dB[i] = (__bf16)sW[i]; }
        }

        const v16bf a0 = load_a_frag<32>(lA[buf], wr * 32,      lane);
        const v16bf a1 = load_a_frag<32>(lA[buf], wr * 32 + 16, lane);
#pragma unroll
        for (int j = 0; j < 4; ++j) {
            const v16bf b = load_b_frag<32>(lB[buf], wc * 64 + j * 16, lane);
            acc[0][j] = WMMA_BF16(a0, b, acc[0][j]);
            acc[1][j] = WMMA_BF16(a1, b, acc[1][j]);
        }
        __syncthreads();
    }

    // C/D layout: lanes 0-15: (M=r, N=lane) ; lanes 16-31: (M=r+8, N=lane-16)
    const int laneN = lane & 15;
    const int mHalf = (lane >> 4) * 8;
#pragma unroll
    for (int i = 0; i < 2; ++i)
#pragma unroll
        for (int j = 0; j < 4; ++j) {
            const int n  = nTile + wc * 64 + j * 16 + laneN;
            const float bv = bias[n];
#pragma unroll
            for (int r = 0; r < 8; ++r) {
                const int m = mTile + wr * 32 + i * 16 + mHalf + r;
                C[(size_t)m * N + n] = acc[i][j][r] + bv;
            }
        }
}

// ---------------------------------------------------------------------------
// Kernel 2: RoPE (optional) + repack [B,S,D] f32 -> head-major [BH,S,DH] bf16
// One thread handles the (dd, dd+32) rotation pair.
// theta_i = 10000^(-(2i)/D)  (reference divides by D=1024, not DH)
// ---------------------------------------------------------------------------
__global__ __launch_bounds__(256) void rope_pack(
    const float* __restrict__ proj, __bf16* __restrict__ dst, int doRope)
{
    const int gid = blockIdx.x * 256 + threadIdx.x;   // BH*S*32 threads
    const int dd  = gid & 31;
    const int s   = (gid >> 5) & (S_ - 1);
    const int bh  = gid >> 16;                        // S_*32 == 1<<16
    const int b   = bh / H_;
    const int h   = bh % H_;

    const size_t src = (size_t)(b * S_ + s) * D_ + h * DH_ + dd;
    const float x1 = proj[src];
    const float x2 = proj[src + 32];
    float y1 = x1, y2 = x2;
    if (doRope) {
        const float freq = __powf(10000.f, -(float)(2 * dd) * (1.f / (float)D_));
        const float ang  = (float)s * freq;
        float sn, cs;
        __sincosf(ang, &sn, &cs);
        y1 = x1 * cs - x2 * sn;   // q*cos + rotate_half(q)*sin, first half
        y2 = x2 * cs + x1 * sn;   // second half
    }
    const size_t d0 = ((size_t)bh * S_ + s) * DH_ + dd;
    dst[d0]      = (__bf16)y1;
    dst[d0 + 32] = (__bf16)y2;
}

// ---------------------------------------------------------------------------
// Kernel 3: scores = qh @ kh^T * (1/sqrt(DH)), fused causal mask, f32 out.
// K=DH=64 fits entirely in LDS: stage full 128x64 Q and K tiles with
// GLOBAL_LOAD_ASYNC_TO_LDS_B128 (pure bf16 copy, ASYNCcnt-tracked), then a
// single barrier and 16 back-to-back WMMAs.
// ---------------------------------------------------------------------------
__global__ __launch_bounds__(256) void attn_scores(
    const __bf16* __restrict__ qh, const __bf16* __restrict__ kh,
    float* __restrict__ attn)
{
    __shared__ __bf16 lA[128 * 64];   // Q tile, row pitch 64
    __shared__ __bf16 lB[128 * 64];   // K tile (N-major), row pitch 64

    const int tid  = threadIdx.x;
    const int lane = tid & 31;
    const int w    = tid >> 5;
    const int wr   = w >> 1, wc = w & 1;
    const int bh   = blockIdx.z;
    const int mTile = blockIdx.y * 128;   // query rows
    const int nTile = blockIdx.x * 128;   // key cols

    // Async-stage both tiles: each thread copies 32 elems (4 x b128) per tile.
    const int row   = tid >> 1;          // 0..127
    const int halfo = (tid & 1) * 32;    // 0 or 32 elements
    const __bf16* srcQ = qh + ((size_t)bh * S_ + mTile + row) * DH_ + halfo;
    const __bf16* srcK = kh + ((size_t)bh * S_ + nTile + row) * DH_ + halfo;
    __bf16* dQ = &lA[row * 64 + halfo];
    __bf16* dK = &lB[row * 64 + halfo];
#pragma unroll
    for (int i = 0; i < 4; ++i) {
        async_b128(dQ + i * 8, srcQ + i * 8);
        async_b128(dK + i * 8, srcK + i * 8);
    }
    wait_async0();
    __syncthreads();

    v8f acc[2][4];
#pragma unroll
    for (int i = 0; i < 2; ++i)
#pragma unroll
        for (int j = 0; j < 4; ++j)
#pragma unroll
            for (int e = 0; e < 8; ++e) acc[i][j][e] = 0.f;

#pragma unroll
    for (int ks = 0; ks < 2; ++ks) {                  // two K=32 slabs
        const __bf16* pA = lA + ks * 32;
        const __bf16* pB = lB + ks * 32;
        const v16bf a0 = load_a_frag<64>(pA, wr * 32,      lane);
        const v16bf a1 = load_a_frag<64>(pA, wr * 32 + 16, lane);
#pragma unroll
        for (int j = 0; j < 4; ++j) {
            const v16bf b = load_b_frag<64>(pB, wc * 64 + j * 16, lane);
            acc[0][j] = WMMA_BF16(a0, b, acc[0][j]);
            acc[1][j] = WMMA_BF16(a1, b, acc[1][j]);
        }
    }

    const int laneN = lane & 15;
    const int mHalf = (lane >> 4) * 8;
    float* out = attn + (size_t)bh * S_ * S_;
#pragma unroll
    for (int i = 0; i < 2; ++i)
#pragma unroll
        for (int j = 0; j < 4; ++j) {
            const int kIdx = nTile + wc * 64 + j * 16 + laneN;
#pragma unroll
            for (int r = 0; r < 8; ++r) {
                const int qIdx = mTile + wr * 32 + i * 16 + mHalf + r;
                const float v  = acc[i][j][r] * 0.125f;    // 1/sqrt(64)
                out[(size_t)qIdx * S_ + kIdx] = (kIdx <= qIdx) ? v : -INFINITY;
            }
        }
}

// ---------------------------------------------------------------------------
// Kernel 4: in-place row softmax. One wave32 per row of 2048 (64 elems/lane).
// exp(-inf - max) == 0 handles masked entries.
// ---------------------------------------------------------------------------
__global__ __launch_bounds__(256) void softmax_rows(float* __restrict__ attn)
{
    const int lane = threadIdx.x & 31;
    const int wid  = threadIdx.x >> 5;
    const size_t row = (size_t)blockIdx.x * 8 + wid;
    float* p = attn + row * S_;

    float x[64];
    float mx = -INFINITY;
#pragma unroll
    for (int i = 0; i < 64; ++i) { x[i] = p[lane + i * 32]; mx = fmaxf(mx, x[i]); }
#pragma unroll
    for (int m = 16; m >= 1; m >>= 1) mx = fmaxf(mx, __shfl_xor(mx, m, 32));
    float sum = 0.f;
#pragma unroll
    for (int i = 0; i < 64; ++i) { x[i] = __expf(x[i] - mx); sum += x[i]; }
#pragma unroll
    for (int m = 16; m >= 1; m >>= 1) sum += __shfl_xor(sum, m, 32);
    const float inv = 1.f / sum;
#pragma unroll
    for (int i = 0; i < 64; ++i) p[lane + i * 32] = x[i] * inv;
}

// ---------------------------------------------------------------------------
// Kernel 5: ctx = attn @ vh per head; M=S, N=DH=64, K=S.
// This is the HBM-roofline kernel (streams the 536MB attn tensor), so the
// LDS tiles are double-buffered: f32 loads + bf16 cvt of slab i+1 overlap
// the WMMAs of slab i. Writes ctx in merged [B,S,D] layout.
// Wave grid 4x2, each wave 32x32 -> 2x2 wmma tiles.
// ---------------------------------------------------------------------------
__global__ __launch_bounds__(256) void attn_v(
    const float* __restrict__ attn, const __bf16* __restrict__ vh,
    float* __restrict__ ctx)
{
    __shared__ __bf16 lA[2][128 * 32];
    __shared__ __bf16 lB[2][64 * 32];

    const int tid  = threadIdx.x;
    const int lane = tid & 31;
    const int w    = tid >> 5;
    const int wr   = w >> 1, wc = w & 1;
    const int bh   = blockIdx.z;
    const int b    = bh / H_;
    const int h    = bh % H_;
    const int mTile = blockIdx.x * 128;

    v8f acc[2][2];
#pragma unroll
    for (int i = 0; i < 2; ++i)
#pragma unroll
        for (int j = 0; j < 2; ++j)
#pragma unroll
            for (int e = 0; e < 8; ++e) acc[i][j][e] = 0.f;

    const int rowA  = tid >> 1;
    const int colbA = (tid & 1) * 16;
    const int nB    = tid & 63;       // 0..63
    const int kqB   = tid >> 6;       // 0..3
    const float*  arow = attn + (size_t)bh * S_ * S_ + (size_t)(mTile + rowA) * S_ + colbA;
    const __bf16* vcol = vh + (size_t)bh * S_ * DH_ + nB;

    // prologue: stage slab 0
    {
        __bf16* dA = &lA[0][rowA * 32 + colbA];
#pragma unroll
        for (int i = 0; i < 16; ++i) dA[i] = (__bf16)arow[i];
#pragma unroll
        for (int i = 0; i < 8; ++i) {
            const int k = kqB * 8 + i;
            lB[0][nB * 32 + k] = vcol[(size_t)k * DH_];
        }
    }
    __syncthreads();

    int buf = 0;
    for (int k0 = 0; k0 < S_; k0 += 32, buf ^= 1) {
        if (k0 + 32 < S_) {          // stage next slab, overlapping WMMAs
            const float* sA = arow + k0 + 32;
            __bf16* dA = &lA[buf ^ 1][rowA * 32 + colbA];
#pragma unroll
            for (int i = 0; i < 16; ++i) dA[i] = (__bf16)sA[i];
#pragma unroll
            for (int i = 0; i < 8; ++i) {
                const int k = kqB * 8 + i;
                lB[buf ^ 1][nB * 32 + k] = vcol[(size_t)(k0 + 32 + k) * DH_];
            }
        }

        const v16bf a0 = load_a_frag<32>(lA[buf], wr * 32,      lane);
        const v16bf a1 = load_a_frag<32>(lA[buf], wr * 32 + 16, lane);
#pragma unroll
        for (int j = 0; j < 2; ++j) {
            const v16bf bfr = load_b_frag<32>(lB[buf], wc * 32 + j * 16, lane);
            acc[0][j] = WMMA_BF16(a0, bfr, acc[0][j]);
            acc[1][j] = WMMA_BF16(a1, bfr, acc[1][j]);
        }
        __syncthreads();
    }

    const int laneN = lane & 15;
    const int mHalf = (lane >> 4) * 8;
#pragma unroll
    for (int i = 0; i < 2; ++i)
#pragma unroll
        for (int j = 0; j < 2; ++j) {
            const int dh = wc * 32 + j * 16 + laneN;
#pragma unroll
            for (int r = 0; r < 8; ++r) {
                const int s = mTile + wr * 32 + i * 16 + mHalf + r;
                ctx[(size_t)b * S_ * D_ + (size_t)s * D_ + h * DH_ + dh] = acc[i][j][r];
            }
        }
}

// ---------------------------------------------------------------------------
// Host-side orchestration
// ---------------------------------------------------------------------------
extern "C" void kernel_launch(void* const* d_in, const int* in_sizes, int n_in,
                              void* d_out, int out_size, void* d_ws, size_t ws_size,
                              hipStream_t stream) {
    const float* Q  = (const float*)d_in[0];
    const float* Km = (const float*)d_in[1];
    const float* V  = (const float*)d_in[2];
    // d_in[3] = valid_lens: lower-triangular causal mask, applied analytically
    const float* Wq = (const float*)d_in[4];
    const float* bq = (const float*)d_in[5];
    const float* Wk = (const float*)d_in[6];
    const float* bk = (const float*)d_in[7];
    const float* Wv = (const float*)d_in[8];
    const float* bv = (const float*)d_in[9];
    const float* Wo = (const float*)d_in[10];
    const float* bo = (const float*)d_in[11];

    float* out  = (float*)d_out;                      // [B,S,D]
    float* attn = out + (size_t)B_ * S_ * D_;         // [BH,S,S]

    const size_t MD = (size_t)B_ * S_ * D_;           // 4096*1024
    float*  qp  = (float*)d_ws;
    float*  kp  = qp + MD;
    float*  vp  = kp + MD;
    float*  ctx = vp + MD;
    __bf16* qh  = (__bf16*)(ctx + MD);
    __bf16* kh  = qh + (size_t)BH_ * S_ * DH_;
    __bf16* vh  = kh + (size_t)BH_ * S_ * DH_;

    const dim3 blk(256);
    const dim3 gGemm(D_ / 128, (B_ * S_) / 128);      // (8, 32)

    gemm_xw_bias<<<gGemm, blk, 0, stream>>>(Q,  Wq, bq, qp, B_ * S_, D_, D_);
    gemm_xw_bias<<<gGemm, blk, 0, stream>>>(Km, Wk, bk, kp, B_ * S_, D_, D_);
    gemm_xw_bias<<<gGemm, blk, 0, stream>>>(V,  Wv, bv, vp, B_ * S_, D_, D_);

    const dim3 gRope((BH_ * S_ * 32) / 256);
    rope_pack<<<gRope, blk, 0, stream>>>(qp, qh, 1);
    rope_pack<<<gRope, blk, 0, stream>>>(kp, kh, 1);
    rope_pack<<<gRope, blk, 0, stream>>>(vp, vh, 0);

    const dim3 gScore(S_ / 128, S_ / 128, BH_);       // (16,16,32)
    attn_scores<<<gScore, blk, 0, stream>>>(qh, kh, attn);

    const dim3 gSm((BH_ * S_) / 8);                   // 8192 blocks
    softmax_rows<<<gSm, blk, 0, stream>>>(attn);

    const dim3 gAV(S_ / 128, 1, BH_);                 // (16,1,32)
    attn_v<<<gAV, blk, 0, stream>>>(attn, vh, ctx);

    gemm_xw_bias<<<gGemm, blk, 0, stream>>>(ctx, Wo, bo, out, B_ * S_, D_, D_);
}
